// GaussianUpsampling_46239617909154
// MI455X (gfx1250) — matrix operand order repaired and checked by reading, not compiled, not run on hardware
//
#include <hip/hip_runtime.h>
#include <hip/hip_bf16.h>

// ---------------------------------------------------------------------------
// Gaussian upsampling attention, MI455X (gfx1250, wave32, WMMA).
//   c[t]      = cumsum(ds)[t] - 0.5*ds[t]
//   e[f,t]    = -0.1*(t_f - c[t])^2  (+ -inf where d_mask==0)
//   p_attn    = softmax_t(e)                       (f32 output)
//   out       = p_attn @ hs                        (f32 output, bf16 WMMA)
// B=16, T_TEXT=512, T_FEATS=4096, ADIM=512
// ---------------------------------------------------------------------------

#define B_     16
#define TT     512      // T_text (tokens)  == K of GEMM
#define TF     4096     // T_feats (frames) == M of GEMM
#define AD     512      // adim             == N of GEMM
#define FTILE  32       // frames per workgroup in fused kernel
#define DELTA  0.1f

typedef __attribute__((ext_vector_type(16))) __bf16 v16bf;
typedef __attribute__((ext_vector_type(8)))  float  v8f;

__device__ __forceinline__ unsigned short f2bf(float f) {
    unsigned int u = __builtin_bit_cast(unsigned int, f);
    u += 0x7FFFu + ((u >> 16) & 1u);           // round-to-nearest-even
    return (unsigned short)(u >> 16);
}

// ---------------------------------------------------------------------------
// Kernel 1: per-batch wave32 inclusive scan -> centers c[b][t]
// ---------------------------------------------------------------------------
__global__ __launch_bounds__(32) void gauss_cumsum(const float* __restrict__ ds,
                                                   float* __restrict__ c) {
    const int b    = blockIdx.x;
    const int lane = threadIdx.x;
    float carry = 0.0f;
    for (int ch = 0; ch < TT / 32; ++ch) {
        const int t = ch * 32 + lane;
        const float d0 = ds[b * TT + t];
        float v = d0;
#pragma unroll
        for (int off = 1; off < 32; off <<= 1) {
            float n = __shfl_up(v, off, 32);
            if (lane >= off) v += n;
        }
        c[b * TT + t] = carry + v - 0.5f * d0;
        carry += __shfl(v, 31, 32);
    }
}

// ---------------------------------------------------------------------------
// Kernel 2: hs (f32, [b][t][d]) -> hsT (bf16, [b][d][t])  (32x32 LDS tiles)
// ---------------------------------------------------------------------------
__global__ __launch_bounds__(256) void gauss_hsT(const float* __restrict__ hs,
                                                 unsigned short* __restrict__ hsT) {
    __shared__ unsigned short tile[32][33];
    const int b  = blockIdx.z;
    const int t0 = blockIdx.y * 32;
    const int d0 = blockIdx.x * 32;
    const int tx = threadIdx.x;          // 0..31
    const int ty = threadIdx.y;          // 0..7
    const float* src = hs + ((size_t)b * TT + t0) * AD + d0;
#pragma unroll
    for (int i = 0; i < 32; i += 8)
        tile[ty + i][tx] = f2bf(src[(size_t)(ty + i) * AD + tx]);
    __syncthreads();
    unsigned short* dst = hsT + ((size_t)b * AD + d0) * TT + t0;
#pragma unroll
    for (int i = 0; i < 32; i += 8)
        dst[(size_t)(ty + i) * TT + tx] = tile[tx][ty + i];
}

// ---------------------------------------------------------------------------
// Kernel 3: fused softmax + bf16-WMMA GEMM.
//   grid = B * (TF/FTILE) workgroups, 256 threads (8 waves of 32).
// ---------------------------------------------------------------------------
__global__ __launch_bounds__(256) void gauss_fused(
    const float* __restrict__ c,
    const unsigned short* __restrict__ hsT,
    const unsigned char* __restrict__ h_masks,
    const unsigned char* __restrict__ d_masks,
    float* __restrict__ out,
    float* __restrict__ p_attn) {

    __shared__ __align__(16) float          c_s[TT];
    __shared__ __align__(16) float          madd_s[TT];         // 0 or -inf
    __shared__ __align__(16) unsigned short p_s[FTILE][520];    // bf16 p tile, padded

    const int bid = blockIdx.x;
    const int b   = bid / (TF / FTILE);
    const int f0  = (bid % (TF / FTILE)) * FTILE;

    const int tid  = threadIdx.x;
    const int lane = tid & 31;
    const int w    = tid >> 5;

    for (int i = tid; i < TT; i += 256) {
        c_s[i]    = c[b * TT + i];
        madd_s[i] = d_masks[b * TT + i] ? 0.0f : -__builtin_inff();
    }
    __syncthreads();

    // ------------------ softmax: 8 waves x 4 iters = 32 rows ------------------
#pragma unroll 1
    for (int it = 0; it < FTILE / 8; ++it) {
        const int fl = w + it * 8;           // local frame row
        const int fg = f0 + fl;              // global frame
        const float tval = h_masks[b * TF + fg] ? (float)fg : 0.0f;

        float e[16];
        float mx = -__builtin_inff();
#pragma unroll
        for (int j = 0; j < 16; ++j) {
            const int t = j * 32 + lane;
            const float d = tval - c_s[t];
            const float en = -DELTA * d * d + madd_s[t];
            e[j] = en;
            mx = fmaxf(mx, en);
        }
#pragma unroll
        for (int off = 16; off > 0; off >>= 1)
            mx = fmaxf(mx, __shfl_xor(mx, off, 32));

        float s = 0.0f;
#pragma unroll
        for (int j = 0; j < 16; ++j) {
            const float p = __expf(e[j] - mx);
            e[j] = p;
            s += p;
        }
#pragma unroll
        for (int off = 16; off > 0; off >>= 1)
            s += __shfl_xor(s, off, 32);
        const float inv = 1.0f / s;

        float* prow = p_attn + ((size_t)b * TF + fg) * TT;
#pragma unroll
        for (int j = 0; j < 16; ++j) {
            const int t = j * 32 + lane;
            const float p = e[j] * inv;
            prow[t] = p;                     // coalesced f32 store
            p_s[fl][t] = f2bf(p);            // bf16 copy for WMMA A
        }
    }
    __syncthreads();

    // ------------------ GEMM: out[32 x 512] = p[32 x 512] @ hs[512 x 512] ----
    // wave tile: 16(M) x 128(N); 2 M-groups x 4 N-groups = 8 waves.
    const int frow = (w & 1) * 16;
    const int dcol = (w >> 1) * 128;
    const int lm = lane & 15;
    const int lh = lane >> 4;

    v8f acc[8];
#pragma unroll
    for (int j = 0; j < 8; ++j)
#pragma unroll
        for (int r = 0; r < 8; ++r) acc[j][r] = 0.0f;

    const unsigned short* hB = hsT + (size_t)b * AD * TT;

#pragma unroll 2
    for (int kc = 0; kc < TT; kc += 32) {
        // A operand (16x32 bf16): lane<16 -> row lm, K kb..kb+7 & kb+16..kb+23
        union { uint4 q[2]; v16bf v; } A;
        const unsigned short* pa = &p_s[frow + lm][kc + lh * 8];
        A.q[0] = *(const uint4*)pa;          // K = kc+kb .. +7
        A.q[1] = *(const uint4*)(pa + 16);   // K = kc+kb+16 .. +23
#pragma unroll
        for (int j = 0; j < 8; ++j) {
            // B operand (32x16 bf16): lane holds K kb..kb+7 & kb+16..kb+23 of column N
            union { uint4 q[2]; v16bf v; } Bv;
            const unsigned short* pb =
                hB + (size_t)(dcol + j * 16 + lm) * TT + kc + lh * 8;
            Bv.q[0] = ((const uint4*)pb)[0];
            Bv.q[1] = ((const uint4*)pb)[2];
            acc[j] = __builtin_amdgcn_wmma_f32_16x16x32_bf16(
                false, A.v, false, Bv.v, (short)0, acc[j], false, false);
        }
    }

    // ------------------ store C tiles (f32) ----------------------------------
    float* ob = out + ((size_t)b * TF + f0 + frow) * AD + dcol;
#pragma unroll
    for (int j = 0; j < 8; ++j) {
#pragma unroll
        for (int r = 0; r < 8; ++r) {
            const int M = r + lh * 8;        // VGPR r: lanes 0-15 M=r, 16-31 M=r+8
            ob[(size_t)M * AD + j * 16 + lm] = acc[j][r];
        }
    }
}

// ---------------------------------------------------------------------------
extern "C" void kernel_launch(void* const* d_in, const int* in_sizes, int n_in,
                              void* d_out, int out_size, void* d_ws, size_t ws_size,
                              hipStream_t stream) {
    (void)in_sizes; (void)n_in; (void)out_size; (void)ws_size;

    const float*         hs      = (const float*)d_in[0];          // (B,TT,AD) f32
    const float*         ds      = (const float*)d_in[1];          // (B,TT)    f32
    const unsigned char* h_masks = (const unsigned char*)d_in[2];  // (B,TF) bool
    const unsigned char* d_masks = (const unsigned char*)d_in[3];  // (B,TT) bool

    float* out_p  = (float*)d_out;                                  // (B,TF,AD)
    float* pattn  = out_p + (size_t)B_ * TF * AD;                   // (B,TF,TT)

    // workspace: hsT bf16 (8 MiB) then c (32 KiB)
    unsigned short* hsT = (unsigned short*)d_ws;
    float*          cws = (float*)((char*)d_ws + (size_t)B_ * AD * TT * sizeof(unsigned short));

    gauss_cumsum<<<B_, 32, 0, stream>>>(ds, cws);

    dim3 gT(AD / 32, TT / 32, B_), bT(32, 8);
    gauss_hsT<<<gT, bT, 0, stream>>>(hs, hsT);

    gauss_fused<<<B_ * (TF / FTILE), 256, 0, stream>>>(cws, hsT, h_masks, d_masks,
                                                       out_p, pattn);
}
